// CrossAttention_67164698575520
// MI455X (gfx1250) — matrix operand order repaired and checked by reading, not compiled, not run on hardware
//
#include <hip/hip_runtime.h>
#include <hip/hip_bf16.h>

// ---------------- problem constants ----------------
#define DIM      5120
#define HEADS    40
#define HEAD_DIM 128
#define TEXT_LEN 512
#define KVP      288          // kv length (257) padded up to multiple of 32
#define KVT16    (KVP/16)     // 18
#define KVT32    (KVP/32)     // 9

// ---------------- CDNA5 WMMA types ----------------
typedef __attribute__((ext_vector_type(16))) __bf16          v16bf;
typedef __attribute__((ext_vector_type(8)))  float           v8f;
typedef __attribute__((ext_vector_type(8)))  unsigned short  u16x8;

struct bfrag { u16x8 lo, hi; };   // 32 bytes == v16bf

__device__ __forceinline__ unsigned short f2bf(float f) {
    unsigned u = __builtin_bit_cast(unsigned, f);
    u = (u + 0x7FFFu + ((u >> 16) & 1u)) >> 16;      // round-to-nearest-even
    return (unsigned short)u;
}
__device__ __forceinline__ float bf2f(unsigned short s) {
    unsigned u = ((unsigned)s) << 16;
    return __builtin_bit_cast(float, u);
}

// D = A(16x32 bf16) * B(32x16 bf16) + C(16x16 f32)
__device__ __forceinline__ v8f wmma_bf16(v16bf a, v16bf b, v8f c) {
    return __builtin_amdgcn_wmma_f32_16x16x32_bf16(false, a, false, b,
                                                   (short)0, c, false, false);
}

// A-matrix 16x32 bf16 fragment (ISA 7.12.2):
//   lanes 0-15 : row M=lane,    K = {0..7, 16..23}
//   lanes16-31 : row M=lane-16, K = {8..15, 24..31}
__device__ __forceinline__ v16bf load_frag_a(const unsigned short* p, int stride, int lane) {
    int half = lane >> 4;
    int r    = lane & 15;
    const unsigned short* row = p + (size_t)r * stride + half * 8;
    bfrag f;
    f.lo = *(const u16x8*)(row);        // K  0.. 7  (+8 if upper half)
    f.hi = *(const u16x8*)(row + 16);   // K 16..23  (+8 if upper half)
    return __builtin_bit_cast(v16bf, f);
}

// B-matrix 32x16 bf16 fragment:
//   lanes 0-15 : col N=lane,    K = {0..7, 16..23}
//   lanes16-31 : col N=lane-16, K = {8..15, 24..31}
__device__ __forceinline__ v16bf load_frag_b(const unsigned short* p, int stride, int lane) {
    int half = lane >> 4;
    int r    = lane & 15;
    const unsigned short* row = p + (size_t)r * stride + half * 8;
    bfrag f;
    f.lo = *(const u16x8*)(row);
    f.hi = *(const u16x8*)(row + 16);
    return __builtin_bit_cast(v16bf, f);
}

// ---------------- utility kernels ----------------
__global__ void zero_u16(unsigned short* p, int n) {
    int i = blockIdx.x * 256 + threadIdx.x;
    if (i < n) p[i] = 0;
}

// vt[h][d][j] = vp[j][h*128+d]   (makes P*V B-operand K-contiguous)
__global__ void transpose_v(const unsigned short* __restrict__ vp,
                            unsigned short* __restrict__ vt) {
    int idx = blockIdx.x * 256 + threadIdx.x;
    if (idx >= HEADS * HEAD_DIM * KVP) return;
    int j = idx % KVP;
    int t = idx / KVP;
    int d = t % HEAD_DIM;
    int h = t / HEAD_DIM;
    vt[idx] = vp[(size_t)j * DIM + h * HEAD_DIM + d];
}

// in-place RMS norm of bf16 rows (f32 math), row = blockIdx.x
__global__ __launch_bounds__(256) void rmsnorm_bf16(unsigned short* __restrict__ x,
                                                    const float* __restrict__ g,
                                                    int cols) {
    __shared__ float red[256];
    unsigned short* xr = x + (size_t)blockIdx.x * cols;
    float s = 0.f;
    for (int c = threadIdx.x; c < cols; c += 256) {
        float v = bf2f(xr[c]);
        s += v * v;
    }
    red[threadIdx.x] = s;
    __syncthreads();
    for (int st = 128; st > 0; st >>= 1) {
        if (threadIdx.x < st) red[threadIdx.x] += red[threadIdx.x + st];
        __syncthreads();
    }
    float scale = rsqrtf(red[0] / (float)cols + 1e-6f);
    for (int c = threadIdx.x; c < cols; c += 256)
        xr[c] = f2bf(bf2f(xr[c]) * scale * g[c]);
}

// ---------------- NT GEMM: C[M,N] = A[M,K] * B[N,K]^T + bias, bf16 out ----------------
// f32 inputs converted to bf16 during the global->LDS stage; f32 accumulate via WMMA.
// 1-D grid decoded into supertiles of GM M-panels x ALL N-panels for L2 blocking,
// ping-pong LDS buffers for one-barrier-per-K-step pipelining.
__global__ __launch_bounds__(256) void gemm_nt_bias_bf16(
        const float* __restrict__ A, const float* __restrict__ B,
        const float* __restrict__ bias, unsigned short* __restrict__ C,
        int M, int N, int K) {
    constexpr int BM = 128, BN = 128, BK = 32, LDT = BK + 8;  // +8 bf16 pad
    constexpr int GM = 16;                                    // M-panels per L2 supertile
    __shared__ unsigned short As[2][BM * LDT];
    __shared__ unsigned short Bs[2][BN * LDT];

    const int tid  = threadIdx.x;
    const int lane = tid & 31;
    const int wid  = tid >> 5;          // 8 waves
    const int wr   = wid >> 1;          // 4 waves along M (32 rows each)
    const int wc   = wid & 1;           // 2 waves along N (64 cols each)

    // ---- L2-friendly supertile decode: GM M-panels x all N-panels per chunk ----
    const int mPanels = (M + BM - 1) / BM;
    const int nPanels = N / BN;
    const int bid     = blockIdx.x;
    const int chunk   = bid / (GM * nPanels);
    const int rlin    = bid - chunk * (GM * nPanels);
    const int cStart  = chunk * GM;
    int gmEff = mPanels - cStart; if (gmEff > GM) gmEff = GM;
    const int m0 = (cStart + (rlin % gmEff)) * BM;
    const int n0 = (rlin / gmEff) * BN;

    v8f acc[2][4];
    #pragma unroll
    for (int i = 0; i < 2; ++i)
        #pragma unroll
        for (int j = 0; j < 4; ++j) acc[i][j] = v8f{};

    const int lrow = tid >> 1;          // 0..127
    const int lseg = (tid & 1) * 16;    // 0 or 16
    int ga = m0 + lrow; if (ga > M - 1) ga = M - 1;
    int gb = n0 + lrow; if (gb > N - 1) gb = N - 1;
    const float* aRow = A + (size_t)ga * K + lseg;
    const float* bRow = B + (size_t)gb * K + lseg;

    // convert 16+16 f32 -> bf16 and stage to LDS buffer `buf`
    auto stage = [&](int buf,
                     float4 a0, float4 a1, float4 a2, float4 a3,
                     float4 b0, float4 b1, float4 b2, float4 b3) {
        u16x8 s0, s1, t0, t1;
        s0[0]=f2bf(a0.x); s0[1]=f2bf(a0.y); s0[2]=f2bf(a0.z); s0[3]=f2bf(a0.w);
        s0[4]=f2bf(a1.x); s0[5]=f2bf(a1.y); s0[6]=f2bf(a1.z); s0[7]=f2bf(a1.w);
        s1[0]=f2bf(a2.x); s1[1]=f2bf(a2.y); s1[2]=f2bf(a2.z); s1[3]=f2bf(a2.w);
        s1[4]=f2bf(a3.x); s1[5]=f2bf(a3.y); s1[6]=f2bf(a3.z); s1[7]=f2bf(a3.w);
        t0[0]=f2bf(b0.x); t0[1]=f2bf(b0.y); t0[2]=f2bf(b0.z); t0[3]=f2bf(b0.w);
        t0[4]=f2bf(b1.x); t0[5]=f2bf(b1.y); t0[6]=f2bf(b1.z); t0[7]=f2bf(b1.w);
        t1[0]=f2bf(b2.x); t1[1]=f2bf(b2.y); t1[2]=f2bf(b2.z); t1[3]=f2bf(b2.w);
        t1[4]=f2bf(b3.x); t1[5]=f2bf(b3.y); t1[6]=f2bf(b3.z); t1[7]=f2bf(b3.w);
        *(u16x8*)&As[buf][lrow * LDT + lseg]     = s0;
        *(u16x8*)&As[buf][lrow * LDT + lseg + 8] = s1;
        *(u16x8*)&Bs[buf][lrow * LDT + lseg]     = t0;
        *(u16x8*)&Bs[buf][lrow * LDT + lseg + 8] = t1;
    };

    // ---- prologue: stage K-tile 0 into buffer 0 ----
    {
        const float4* ap = (const float4*)(aRow);
        const float4* bp = (const float4*)(bRow);
        stage(0, ap[0], ap[1], ap[2], ap[3], bp[0], bp[1], bp[2], bp[3]);
    }
    __syncthreads();

    // ---- main loop: compute on cur buffer, load+stage nxt, single barrier ----
    for (int k0 = 0; k0 < K; k0 += BK) {
        const int  cur  = (k0 / BK) & 1;
        const int  nxt  = cur ^ 1;
        const bool more = (k0 + BK) < K;

        float4 a0, a1, a2, a3, b0, b1, b2, b3;
        if (more) {
            const float4* ap = (const float4*)(aRow + k0 + BK);
            const float4* bp = (const float4*)(bRow + k0 + BK);
            a0 = ap[0]; a1 = ap[1]; a2 = ap[2]; a3 = ap[3];
            b0 = bp[0]; b1 = bp[1]; b2 = bp[2]; b3 = bp[3];
            if (k0 + 2 * BK < K) {                   // global_prefetch_b8
                __builtin_prefetch(aRow + k0 + 2 * BK, 0, 1);
                __builtin_prefetch(bRow + k0 + 2 * BK, 0, 1);
            }
        }

        // 8 WMMAs on the current buffer
        v16bf fa0 = load_frag_a(&As[cur][(wr * 32) * LDT],      LDT, lane);
        v16bf fa1 = load_frag_a(&As[cur][(wr * 32 + 16) * LDT], LDT, lane);
        #pragma unroll
        for (int wn = 0; wn < 4; ++wn) {
            v16bf fb = load_frag_b(&Bs[cur][(wc * 64 + wn * 16) * LDT], LDT, lane);
            acc[0][wn] = wmma_bf16(fa0, fb, acc[0][wn]);
            acc[1][wn] = wmma_bf16(fa1, fb, acc[1][wn]);
        }

        if (more)
            stage(nxt, a0, a1, a2, a3, b0, b1, b2, b3);
        __syncthreads();
    }

    // ---- epilogue: bias add, bf16 store (C layout: vgpr r -> M=r(+8), N=lane&15) ----
    const int half = lane >> 4, nn = lane & 15;
    #pragma unroll
    for (int wm = 0; wm < 2; ++wm)
        #pragma unroll
        for (int wn = 0; wn < 4; ++wn) {
            int col = n0 + wc * 64 + wn * 16 + nn;
            float bv = bias[col];
            #pragma unroll
            for (int r = 0; r < 8; ++r) {
                int row = m0 + wr * 32 + wm * 16 + half * 8 + r;
                if (row < M)
                    C[(size_t)row * N + col] = f2bf(acc[wm][wn][r] + bv);
            }
        }
}

// ---------------- attention: one wave per (16 q-rows, head) ----------------
__global__ __launch_bounds__(32) void attn_kernel(
        const unsigned short* __restrict__ qb,   // [S][DIM] bf16 (normed)
        const unsigned short* __restrict__ kp,   // [KVP][DIM] bf16 (normed, padded w/ 0)
        const unsigned short* __restrict__ vt,   // [HEADS][HEAD_DIM][KVP] bf16
        float* __restrict__ out,                 // [S][DIM] f32
        int kvLen) {
    __shared__ float          sc[16 * KVP];
    __shared__ unsigned short P[16 * KVP];

    const int lane = threadIdx.x;
    const int q0   = blockIdx.x * 16;
    const int h    = blockIdx.y;
    const int half = lane >> 4, nn = lane & 15;
    const float scale = 0.088388347648318447f;   // 1/sqrt(128)

    // resident Q fragments: 16 x 128 = 4 x (16x32)
    v16bf qf[4];
    #pragma unroll
    for (int kk = 0; kk < 4; ++kk)
        qf[kk] = load_frag_a(qb + (size_t)q0 * DIM + h * HEAD_DIM + kk * 32, DIM, lane);

    // S = Q * K^T  (18 kv-tiles x 4 k-steps of WMMA)
    for (int jt = 0; jt < KVT16; ++jt) {
        v8f a = v8f{};
        #pragma unroll
        for (int kk = 0; kk < 4; ++kk) {
            v16bf kf = load_frag_b(kp + (size_t)(jt * 16) * DIM + h * HEAD_DIM + kk * 32,
                                   DIM, lane);
            a = wmma_bf16(qf[kk], kf, a);
        }
        #pragma unroll
        for (int r = 0; r < 8; ++r)
            sc[(half * 8 + r) * KVP + jt * 16 + nn] = a[r] * scale;
    }
    __syncthreads();

    // masked softmax (f32), P in bf16; lanes 0..15 each own one row
    if (lane < 16) {
        float* srow = &sc[lane * KVP];
        float mx = -1e30f;
        for (int j = 0; j < kvLen; ++j) mx = fmaxf(mx, srow[j]);
        float sum = 0.f;
        for (int j = 0; j < kvLen; ++j) { float e = __expf(srow[j] - mx); srow[j] = e; sum += e; }
        float inv = 1.0f / sum;
        unsigned short* pr = &P[lane * KVP];
        for (int j = 0; j < kvLen; ++j) pr[j] = f2bf(srow[j] * inv);
        for (int j = kvLen; j < KVP; ++j) pr[j] = 0;
    }
    __syncthreads();

    // O = P * V  (8 n-tiles x 9 k-steps)
    #pragma unroll
    for (int nt = 0; nt < 8; ++nt) {
        v8f a = v8f{};
        #pragma unroll
        for (int kt = 0; kt < KVT32; ++kt) {
            v16bf pf = load_frag_a(&P[kt * 32], KVP, lane);
            v16bf vf = load_frag_b(vt + ((size_t)h * HEAD_DIM + nt * 16) * KVP + kt * 32,
                                   KVP, lane);
            a = wmma_bf16(pf, vf, a);
        }
        #pragma unroll
        for (int r = 0; r < 8; ++r)
            out[(size_t)(q0 + half * 8 + r) * DIM + h * HEAD_DIM + nt * 16 + nn] = a[r];
    }
}

// ---------------- host launcher ----------------
extern "C" void kernel_launch(void* const* d_in, const int* in_sizes, int n_in,
                              void* d_out, int out_size, void* d_ws, size_t ws_size,
                              hipStream_t stream) {
    const float* hs  = (const float*)d_in[0];
    const float* enc = (const float*)d_in[1];
    const float* wq  = (const float*)d_in[2];
    const float* bq  = (const float*)d_in[3];
    const float* wk  = (const float*)d_in[4];
    const float* bk  = (const float*)d_in[5];
    const float* wv  = (const float*)d_in[6];
    const float* bv  = (const float*)d_in[7];
    const float* gq  = (const float*)d_in[8];
    const float* gk  = (const float*)d_in[9];
    float* out = (float*)d_out;

    const int S      = in_sizes[0] / DIM;    // 16384
    const int encLen = in_sizes[1] / DIM;    // 769
    const int kvLen  = encLen - TEXT_LEN;    // 257

    // workspace: q bf16 | k padded bf16 | v padded bf16 | v transposed bf16
    unsigned short* qb = (unsigned short*)d_ws;
    unsigned short* kp = qb + (size_t)S * DIM;
    unsigned short* vp = kp + (size_t)KVP * DIM;
    unsigned short* vt = vp + (size_t)KVP * DIM;

    {   // zero kv pads (rows kvLen..KVP-1 must be zero)
        int n = KVP * DIM;
        int b = (n + 255) / 256;
        zero_u16<<<b, 256, 0, stream>>>(kp, n);
        zero_u16<<<b, 256, 0, stream>>>(vp, n);
    }

    // projections (NT GEMMs, bias fused, bf16 out); 1-D grids, supertile-decoded
    {
        int mP = S / 128, nP = DIM / 128;
        gemm_nt_bias_bf16<<<dim3(mP * nP), 256, 0, stream>>>(hs, wq, bq, qb, S, DIM, DIM);
    }
    {
        int mP = (kvLen + 127) / 128, nP = DIM / 128;
        gemm_nt_bias_bf16<<<dim3(mP * nP), 256, 0, stream>>>(enc, wk, bk, kp, kvLen, DIM, DIM);
        gemm_nt_bias_bf16<<<dim3(mP * nP), 256, 0, stream>>>(enc, wv, bv, vp, kvLen, DIM, DIM);
    }

    // RMS norms
    rmsnorm_bf16<<<S, 256, 0, stream>>>(qb, gq, DIM);
    rmsnorm_bf16<<<kvLen, 256, 0, stream>>>(kp, gk, DIM);

    // per-head V transpose for K-contiguous P*V operand
    {
        int n = HEADS * HEAD_DIM * KVP;
        transpose_v<<<(n + 255) / 256, 256, 0, stream>>>(vp, vt);
    }

    // attention
    dim3 gridA(S / 16, HEADS);
    attn_kernel<<<gridA, 32, 0, stream>>>(qb, kp, vt, out, kvLen);
}
